// MoELayer_85495618994889
// MI455X (gfx1250) — matrix operand order repaired and checked by reading, not compiled, not run on hardware
//
#include <hip/hip_runtime.h>
#include <hip/hip_bf16.h>

// Problem constants (from the reference)
#define T_TOK 8192
#define DDIM  512
#define HDIM  1024
#define NEXP  8
#define TOPK  2

#define KB1 (DDIM / 32)   // k-blocks of W1 (K = D)
#define KB2 (HDIM / 32)   // k-blocks of W2 (K = H)

typedef __attribute__((ext_vector_type(16))) __bf16 v16bf;
typedef __attribute__((ext_vector_type(8)))  float  v8f;
typedef __attribute__((ext_vector_type(4)))  int    v4i;

// CDNA5 async global->LDS copy (ASYNCcnt-tracked), if this toolchain has it.
#if defined(__AMDGCN__) && __has_builtin(__builtin_amdgcn_global_load_async_to_lds_b128)
#define HAVE_ASYNC_LDS 1
#endif

// ---------------------------------------------------------------------------
// WMMA helper (CDNA5: V_WMMA_F32_16X16X32_BF16, wave32, 16x16 C/D in 8 VGPRs)
// ---------------------------------------------------------------------------
__device__ __forceinline__ v8f wmma_bf16(v16bf a, v16bf b, v8f c) {
  // (neg_a, A, neg_b, B, c_mod, C, reuse_a, reuse_b)
  return __builtin_amdgcn_wmma_f32_16x16x32_bf16(false, a, false, b,
                                                 (short)0, c, false, false);
}

// A-matrix 16x32 bf16 fragment from LDS (row-major tile, given row stride).
// ISA layout: lanes 0-15 -> M=lane, K={0..7,16..23}; lanes 16-31 -> M=lane-16,
// K={8..15,24..31}. elem[i<8] = K=koff+i ; elem[i>=8] = K=16+koff+(i-8).
__device__ __forceinline__ v16bf load_a_lds(const __bf16* base, int stride, int lane) {
  int m    = lane & 15;
  int koff = (lane >> 4) << 3;          // 0 or 8
  const __bf16* p0 = base + m * stride + koff;
  v16bf a;
#pragma unroll
  for (int i = 0; i < 8; ++i) a[i] = p0[i];
#pragma unroll
  for (int i = 0; i < 8; ++i) a[8 + i] = p0[16 + i];
  return a;
}

// B-matrix 32x16 fragment from the pre-packed bf16 weights.
// Packed layout: P[((kblk*N + n)<<5) + half*16 + i] = W[kblk*32 + (i<8 ? 8*half+i : 16+8*half+i-8)][n].
// Lane (n = n0+lane&15, half = lane>>4) reads 16 contiguous bf16 (32 B,
// 32-B aligned) -> two global_load_b128, no converts in the hot loop.
__device__ __forceinline__ v16bf load_b_pk(const __bf16* __restrict__ P, int N,
                                           int kblk, int n0, int lane) {
  int n = n0 + (lane & 15);
  const __bf16* p = P + (((size_t)kblk * N + n) << 5) + ((lane >> 4) << 4);
  return *(const v16bf*)p;
}

// ---------------------------------------------------------------------------
// Kernel 0: zero output + counters (harness poisons d_out/d_ws with 0xAA)
// ---------------------------------------------------------------------------
__global__ void init_kernel(float* __restrict__ out, int* __restrict__ counts,
                            float* __restrict__ prob_sums, int n) {
  int i = blockIdx.x * blockDim.x + threadIdx.x;
  if (i < n) out[i] = 0.0f;
  if (i < NEXP) { counts[i] = 0; prob_sums[i] = 0.0f; }
}

// ---------------------------------------------------------------------------
// Kernel 0a: x (f32) -> xbf (bf16), row-major.  One pass; every token row is
// gathered ~TOPK times by expert tiles, so convert once here instead.
// ---------------------------------------------------------------------------
__global__ __launch_bounds__(256)
void cvt_x_kernel(const float* __restrict__ x, __bf16* __restrict__ xbf, long n) {
  long i = ((long)blockIdx.x * blockDim.x + threadIdx.x) * 4;
  if (i + 3 < n) {
    float4 v = *(const float4*)(x + i);
    xbf[i]     = (__bf16)v.x;
    xbf[i + 1] = (__bf16)v.y;
    xbf[i + 2] = (__bf16)v.z;
    xbf[i + 3] = (__bf16)v.w;
  }
}

// ---------------------------------------------------------------------------
// Kernel 0b: pre-pack a weight tensor W[E][K][N] (f32 row-major) into the
// bf16 B-fragment layout  P[e][kblk][n][half][i]  (see load_b_pk).
// ---------------------------------------------------------------------------
__global__ __launch_bounds__(256)
void pack_kernel(const float* __restrict__ W, __bf16* __restrict__ P,
                 int Ktot, int N, long total) {
  long idx = (long)blockIdx.x * blockDim.x + threadIdx.x;
  if (idx >= total) return;
  int i    = (int)(idx & 15);
  int half = (int)((idx >> 4) & 1);
  long r   = idx >> 5;
  int n    = (int)(r % N);
  long r2  = r / N;
  int kblk = (int)(r2 % (Ktot / 32));
  int e    = (int)(r2 / (Ktot / 32));
  int k    = kblk * 32 + (i < 8 ? half * 8 + i : 16 + half * 8 + (i - 8));
  P[idx] = (__bf16)W[((size_t)e * Ktot + k) * N + n];
}

// ---------------------------------------------------------------------------
// Kernel 1: gating — logits, softmax prob sums (loss), top-2 weights, routing
// ---------------------------------------------------------------------------
__global__ __launch_bounds__(256)
void gate_kernel(const float* __restrict__ x, const float* __restrict__ Wg,
                 const float* __restrict__ bg, float* __restrict__ wte,
                 float* __restrict__ prob_sums, int* __restrict__ counts,
                 int* __restrict__ lists) {
  __shared__ float psum[NEXP];
  const int tid = threadIdx.x;
  if (tid < NEXP) psum[tid] = 0.0f;
  __syncthreads();

  const int t = blockIdx.x * blockDim.x + tid;   // T divisible by 256
  float acc[NEXP];
#pragma unroll
  for (int e = 0; e < NEXP; ++e) acc[e] = bg[e];

  const float* xr = x + (size_t)t * DDIM;
  for (int d = 0; d < DDIM; ++d) {
    float xv = xr[d];
    const float* wg = Wg + d * NEXP;   // E contiguous -> broadcast-friendly
#pragma unroll
    for (int e = 0; e < NEXP; ++e) acc[e] += xv * wg[e];
  }

  // full softmax (for load-balancing loss), reduced in LDS first
  float mx = acc[0];
#pragma unroll
  for (int e = 1; e < NEXP; ++e) mx = fmaxf(mx, acc[e]);
  float ex[NEXP], s = 0.0f;
#pragma unroll
  for (int e = 0; e < NEXP; ++e) { ex[e] = __expf(acc[e] - mx); s += ex[e]; }
  float inv = 1.0f / s;
#pragma unroll
  for (int e = 0; e < NEXP; ++e) atomicAdd(&psum[e], ex[e] * inv);

  // top-2 (first-max tie-break matches lax.top_k ordering)
  int i0 = 0;
#pragma unroll
  for (int e = 1; e < NEXP; ++e) if (acc[e] > acc[i0]) i0 = e;
  int i1 = (i0 == 0) ? 1 : 0;
#pragma unroll
  for (int e = 0; e < NEXP; ++e) if (e != i0 && acc[e] > acc[i1]) i1 = e;

  float w0 = 1.0f / (1.0f + __expf(acc[i1] - acc[i0]));  // softmax over top-2
  float w1 = 1.0f - w0;

#pragma unroll
  for (int e = 0; e < NEXP; ++e)
    wte[(size_t)t * NEXP + e] = (e == i0) ? w0 : ((e == i1) ? w1 : 0.0f);

  int p0 = atomicAdd(&counts[i0], 1); lists[i0 * T_TOK + p0] = t;
  int p1 = atomicAdd(&counts[i1], 1); lists[i1 * T_TOK + p1] = t;

  __syncthreads();
  if (tid < NEXP) atomicAdd(&prob_sums[tid], psum[tid]);
}

// ---------------------------------------------------------------------------
// Kernel 2: routed fused FFN.  Block = (expert e, tile of 32 gathered tokens).
// Stage 1: h = relu(x_tile @ W1_e + b1)  (bf16 WMMA, f32 accum, 256-wide slabs)
// Stage 2: y += h @ W2_e                 (bf16 WMMA, f32 accum in registers)
// Epilogue: out[token] += w_te * (y + b2) via global f32 atomics.
// ---------------------------------------------------------------------------
#define XS_LD (DDIM + 8)   // 1040 B row pitch (16-B multiple)
#define HS_LD (256 + 8)    // 528 B row pitch  (16-B multiple)

__global__ __launch_bounds__(256)
void moe_ffn_kernel(const __bf16* __restrict__ xbf,
                    const __bf16* __restrict__ W1p, const float* __restrict__ b1,
                    const __bf16* __restrict__ W2p, const float* __restrict__ b2,
                    const float* __restrict__ wte,
                    const int* __restrict__ counts, const int* __restrict__ lists,
                    float* __restrict__ out) {
  __shared__ __attribute__((aligned(16))) __bf16 xs[32][XS_LD]; // token rows
  __shared__ __attribute__((aligned(16))) __bf16 hs[32][HS_LD]; // relu(h) slab
  __shared__ int    toks_sh[32];
  __shared__ float  wsc_sh[32];

  const int e     = blockIdx.x;
  const int cnt   = counts[e];
  const int tile0 = blockIdx.y * 32;
  if (tile0 >= cnt) return;          // uniform early exit: EXEC stays all-ones

  const int tid  = threadIdx.x;
  const int wave = tid >> 5;
  const int lane = tid & 31;

  if (tid < 32) {
    int idx = tile0 + tid;
    int tok = 0; float sc = 0.0f;
    if (idx < cnt) { tok = lists[e * T_TOK + idx]; sc = wte[(size_t)tok * NEXP + e]; }
    toks_sh[tid] = tok;
    wsc_sh[tid]  = sc;
  }
  __syncthreads();

  // gather 32 token rows of bf16 x into LDS as 16-byte chunks (64 per row)
#ifdef HAVE_ASYNC_LDS
  for (int i = tid; i < 32 * (DDIM / 8); i += 256) {
    int r = i >> 6, c = i & 63;
    const __bf16* g = xbf + (size_t)toks_sh[r] * DDIM + c * 8;
    __builtin_amdgcn_global_load_async_to_lds_b128(
        (v4i*)(void*)g, (v4i*)(void*)&xs[r][c * 8], 0, 0);
  }
  asm volatile("s_wait_asynccnt 0" ::: "memory");
  __syncthreads();
#else
  for (int i = tid; i < 32 * (DDIM / 8); i += 256) {
    int r = i >> 6, c = i & 63;
    const uint4* g = (const uint4*)(xbf + (size_t)toks_sh[r] * DDIM) + c;
    *(uint4*)&xs[r][c * 8] = *g;
  }
  __syncthreads();
#endif

  const __bf16* W1e = W1p + (size_t)e * KB1 * HDIM * 32;  // packed fragments
  const __bf16* W2e = W2p + (size_t)e * KB2 * DDIM * 32;
  const float*  b1e = b1 + e * HDIM;
  const float*  b2e = b2 + e * DDIM;

  const v8f vzero = {0.f, 0.f, 0.f, 0.f, 0.f, 0.f, 0.f, 0.f};

  // y accumulators: M=32 (2 row tiles) x N=64 per wave (4 col tiles)
  v8f y[2][4];
#pragma unroll
  for (int rt = 0; rt < 2; ++rt)
#pragma unroll
    for (int ct = 0; ct < 4; ++ct) y[rt][ct] = vzero;

  const int nout0 = wave * 64;       // this wave's output column base

  for (int it = 0; it < HDIM / 256; ++it) {
    const int hbase = it * 256;
    const int hw0   = hbase + wave * 32;   // this wave's 32 h-columns (global)

    // ---- stage 1: h slab ----
    v8f hacc[2][2];
#pragma unroll
    for (int rt = 0; rt < 2; ++rt)
#pragma unroll
      for (int ct = 0; ct < 2; ++ct) hacc[rt][ct] = vzero;

    for (int kd = 0; kd < DDIM; kd += 32) {
      v16bf a0 = load_a_lds(&xs[0][kd],  XS_LD, lane);
      v16bf a1 = load_a_lds(&xs[16][kd], XS_LD, lane);
      if (kd + 32 < DDIM)
        __builtin_prefetch(W1e + ((((size_t)(kd >> 5) + 1) * HDIM + hw0 + lane) << 5), 0, 1);
#pragma unroll
      for (int ct = 0; ct < 2; ++ct) {
        v16bf b = load_b_pk(W1e, HDIM, kd >> 5, hw0 + ct * 16, lane);
        hacc[0][ct] = wmma_bf16(a0, b, hacc[0][ct]);
        hacc[1][ct] = wmma_bf16(a1, b, hacc[1][ct]);
      }
    }

    // bias + relu + bf16 -> LDS (C layout: lane -> N, vgpr r -> M)
    {
      const int mhi  = (lane >> 4) * 8;
      const int ncol = lane & 15;
#pragma unroll
      for (int ct = 0; ct < 2; ++ct) {
        float bias = b1e[hw0 + ct * 16 + ncol];
        int lcol   = wave * 32 + ct * 16 + ncol;
#pragma unroll
        for (int rt = 0; rt < 2; ++rt)
#pragma unroll
          for (int r = 0; r < 8; ++r) {
            float v = hacc[rt][ct][r] + bias;
            v = v > 0.0f ? v : 0.0f;
            hs[rt * 16 + mhi + r][lcol] = (__bf16)v;
          }
      }
    }
    __syncthreads();

    // ---- stage 2: y += h_slab @ W2 slab ----
    for (int kh = 0; kh < 256; kh += 32) {
      v16bf a0 = load_a_lds(&hs[0][kh],  HS_LD, lane);
      v16bf a1 = load_a_lds(&hs[16][kh], HS_LD, lane);
      if (kh + 32 < 256)
        __builtin_prefetch(W2e + ((((size_t)((hbase + kh) >> 5) + 1) * DDIM + nout0 + lane) << 5), 0, 1);
#pragma unroll
      for (int ct = 0; ct < 4; ++ct) {
        v16bf b = load_b_pk(W2e, DDIM, (hbase + kh) >> 5, nout0 + ct * 16, lane);
        y[0][ct] = wmma_bf16(a0, b, y[0][ct]);
        y[1][ct] = wmma_bf16(a1, b, y[1][ct]);
      }
    }
    __syncthreads();   // before next slab overwrites hs
  }

  // ---- epilogue: out[token] += w_te * (y + b2) ----
  {
    const int mhi  = (lane >> 4) * 8;
    const int ncol = lane & 15;
#pragma unroll
    for (int rt = 0; rt < 2; ++rt)
#pragma unroll
      for (int ct = 0; ct < 4; ++ct) {
        int col     = nout0 + ct * 16 + ncol;
        float bias2 = b2e[col];
#pragma unroll
        for (int r = 0; r < 8; ++r) {
          int m    = rt * 16 + mhi + r;
          float vv = (y[rt][ct][r] + bias2) * wsc_sh[m];
          atomicAdd(&out[(size_t)toks_sh[m] * DDIM + col], vv);
        }
      }
  }
}

// ---------------------------------------------------------------------------
// Kernel 3: load-balancing loss = E * sum_e mean(prob)_e * mean(mask)_e
// ---------------------------------------------------------------------------
__global__ void loss_kernel(const float* __restrict__ prob_sums,
                            const int* __restrict__ counts,
                            float* __restrict__ out) {
  if (blockIdx.x == 0 && threadIdx.x == 0) {
    float s = 0.0f;
#pragma unroll
    for (int e = 0; e < NEXP; ++e)
      s += (prob_sums[e] / (float)T_TOK) * ((float)counts[e] / (float)T_TOK);
    out[(size_t)T_TOK * DDIM] = (float)NEXP * s;
  }
}

// ---------------------------------------------------------------------------
// Launch
// ---------------------------------------------------------------------------
extern "C" void kernel_launch(void* const* d_in, const int* in_sizes, int n_in,
                              void* d_out, int out_size, void* d_ws, size_t ws_size,
                              hipStream_t stream) {
  const float* x  = (const float*)d_in[0];
  const float* W1 = (const float*)d_in[1];
  const float* b1 = (const float*)d_in[2];
  const float* W2 = (const float*)d_in[3];
  const float* b2 = (const float*)d_in[4];
  const float* Wg = (const float*)d_in[5];
  const float* bg = (const float*)d_in[6];
  // d_in[7] = top_k (constexpr TOPK=2 here)
  float* out = (float*)d_out;

  // workspace layout (all 32-B aligned by construction)
  float*  wte       = (float*)d_ws;                       // T*E combine weights
  float*  prob_sums = wte + (size_t)T_TOK * NEXP;         // E
  int*    counts    = (int*)(prob_sums + NEXP);           // E
  int*    lists     = counts + NEXP;                      // E*T token ids
  __bf16* W1p       = (__bf16*)(lists + (size_t)NEXP * T_TOK);
  __bf16* W2p       = W1p + (size_t)NEXP * DDIM * HDIM;   // 8 MB each
  __bf16* xbf       = W2p + (size_t)NEXP * HDIM * DDIM;   // 8 MB

  const long nW = (long)NEXP * DDIM * HDIM;  // same element count for W1 and W2
  const long nX = (long)T_TOK * DDIM;

  const int n_out = T_TOK * DDIM + 1;
  init_kernel<<<(n_out + 255) / 256, 256, 0, stream>>>(out, counts, prob_sums, n_out);

  cvt_x_kernel<<<(int)((nX / 4 + 255) / 256), 256, 0, stream>>>(x, xbf, nX);
  pack_kernel<<<(int)((nW + 255) / 256), 256, 0, stream>>>(W1, W1p, DDIM, HDIM, nW);
  pack_kernel<<<(int)((nW + 255) / 256), 256, 0, stream>>>(W2, W2p, HDIM, DDIM, nW);

  gate_kernel<<<(T_TOK + 255) / 256, 256, 0, stream>>>(x, Wg, bg, wte, prob_sums,
                                                       counts, lists);

  dim3 grid(NEXP, (T_TOK + 31) / 32);
  moe_ffn_kernel<<<grid, 256, 0, stream>>>(xbf, W1p, b1, W2p, b2, wte, counts, lists, out);

  loss_kernel<<<1, 32, 0, stream>>>(prob_sums, counts, out);
}
